// Ember_model_2044404433015
// MI455X (gfx1250) — matrix operand order repaired and checked by reading, not compiled, not run on hardware
//
#include <hip/hip_runtime.h>

typedef __attribute__((ext_vector_type(16))) __bf16 v16bf;
typedef __attribute__((ext_vector_type(8)))  float  v8f;

#define DIN 2381
#define BROWS 65536
#define BN_EPS 1e-5f

// Packed-weight geometry: [kstep][ntile][lane(32)][16 x bf16]  (512 bf16 per (kstep,ntile))
#define W0_KSTEPS 75   // ceil(2381/32)
#define W0_NT 8
#define W1_KSTEPS 4
#define W1_NT 4
#define W2_KSTEPS 2
#define W2_NT 2
#define W3_KSTEPS 1
#define W3_NT 1
#define W0P_ELEMS (W0_KSTEPS * W0_NT * 512)   // 307200
#define W1P_ELEMS (W1_KSTEPS * W1_NT * 512)   // 8192
#define W2P_ELEMS (W2_KSTEPS * W2_NT * 512)   // 2048
#define W3P_ELEMS (W3_KSTEPS * W3_NT * 512)   // 512

__device__ __forceinline__ float ldnt(const float* p) {
    return __builtin_nontemporal_load(p);
}

__device__ __forceinline__ float act_on_f(float h, bool msk, float dv, float alpha, float beta) {
    float delta = msk ? (h - dv) : 0.0f;
    bool upd = msk && (fabsf(delta) < beta);
    return upd ? fmaf(-alpha, delta, h) : h;
}

// Convert a row-major fp32 weight [dout, din] into bf16 WMMA B-matrix lane order:
// dst[((ks*ntiles + nt)*32 + lane)*16 + i] = W[nt*16 + (lane&15)][ks*32 + (i>>3)*16 + (lane>>4)*8 + (i&7)]
// K zero-padded to ksteps*32.
__global__ __launch_bounds__(256) void pack_b_kernel(const float* __restrict__ W,
                                                     __bf16* __restrict__ dst,
                                                     int din, int ntiles, int ksteps)
{
    int idx = blockIdx.x * 256 + threadIdx.x;
    int total = ksteps * ntiles * 512;
    if (idx >= total) return;
    int i    = idx & 15;
    int lane = (idx >> 4) & 31;
    int nt   = (idx >> 9) % ntiles;
    int ks   = idx / (512 * ntiles);
    int n = nt * 16 + (lane & 15);
    int k = ks * 32 + (i >> 3) * 16 + (lane >> 4) * 8 + (i & 7);
    float v = (k < din) ? W[(size_t)n * din + k] : 0.0f;
    dst[idx] = (__bf16)v;
}

__global__ __launch_bounds__(256, 1) void fused_mlp(
    const float* __restrict__ x,
    const __bf16* __restrict__ W0p, const float* __restrict__ b0,
    const __bf16* __restrict__ W1p, const float* __restrict__ b1,
    const __bf16* __restrict__ W2p, const float* __restrict__ b2,
    const __bf16* __restrict__ W3p, const float* __restrict__ b3,
    const float* __restrict__ W4, const float* __restrict__ b4,
    const float* __restrict__ g2, const float* __restrict__ be2,
    const float* __restrict__ m2, const float* __restrict__ v2,
    const float* __restrict__ g3, const float* __restrict__ be3,
    const float* __restrict__ m3, const float* __restrict__ v3,
    const float* __restrict__ g4, const float* __restrict__ be4,
    const float* __restrict__ m4, const float* __restrict__ v4,
    const float* __restrict__ distr, const float* __restrict__ betas,
    const float* __restrict__ alpha_p,
    float* __restrict__ out)
{
    // Per-layer activation staging (bf16). Each wave touches only its own 16 rows
    // of every buffer, so no workgroup barriers are required.
    __shared__ __bf16 sh0[128 * 128];  // 32 KB
    __shared__ __bf16 sh1[128 * 64];   // 16 KB
    __shared__ __bf16 sh2[128 * 32];   //  8 KB

    const int tid  = threadIdx.x;
    const int wave = tid >> 5;
    const int lane = tid & 31;
    const int half = lane >> 4;     // 0: lanes 0-15, 1: lanes 16-31
    const int lm   = lane & 15;
    const int rowBase = blockIdx.x * 128 + wave * 16;
    const float alpha = alpha_p[0];

    const v8f vzero = {0.f, 0.f, 0.f, 0.f, 0.f, 0.f, 0.f, 0.f};
    const int kA0 = half * 8;       // 16-bit A layout: K half-offset per lane group

    // ================= Layer 0: [16,2381] x [2381,128] (bf16 WMMA, f32 acc) =================
    v8f acc[8];
#pragma unroll
    for (int i = 0; i < 8; i++) acc[i] = vzero;

    const float* xrow = x + (size_t)(rowBase + lm) * DIN;   // A: M = lane&15
    const __bf16* w0l = W0p + lane * 16;                    // B: per-lane packed base

    // Software pipeline: fp32 A-chunk for step ks is prefetched during step ks-1.
    float fa[16];
#pragma unroll
    for (int i = 0; i < 8; i++) fa[i]     = ldnt(xrow + kA0 + i);
#pragma unroll
    for (int i = 0; i < 8; i++) fa[8 + i] = ldnt(xrow + kA0 + 16 + i);

    for (int ks = 0; ks < W0_KSTEPS; ks++) {
        v16bf a;
#pragma unroll
        for (int i = 0; i < 16; i++) a[i] = (__bf16)fa[i];

        // Prefetch next A chunk (guarded when it is the K tail)
        if (ks + 2 < W0_KSTEPS) {
            const int kn = (ks + 1) * 32;
#pragma unroll
            for (int i = 0; i < 8; i++) fa[i]     = ldnt(xrow + kn + kA0 + i);
#pragma unroll
            for (int i = 0; i < 8; i++) fa[8 + i] = ldnt(xrow + kn + kA0 + 16 + i);
        } else if (ks + 2 == W0_KSTEPS) {
            const int kn = (ks + 1) * 32;   // tail chunk: zero-pad past DIN
#pragma unroll
            for (int i = 0; i < 8; i++) { int k = kn + kA0 + i;      fa[i]     = (k < DIN) ? ldnt(xrow + k) : 0.0f; }
#pragma unroll
            for (int i = 0; i < 8; i++) { int k = kn + kA0 + 16 + i; fa[8 + i] = (k < DIN) ? ldnt(xrow + k) : 0.0f; }
        }

        // B tiles in batches of 4: loads issue as a clause, overlap with WMMAs
        const v16bf* wk = (const v16bf*)(w0l + (size_t)ks * (W0_NT * 512));
        v16bf bm[4];
#pragma unroll
        for (int nt = 0; nt < 4; nt++) bm[nt] = wk[nt * 32];   // nt*512 bf16
#pragma unroll
        for (int nt = 0; nt < 4; nt++)
            acc[nt] = __builtin_amdgcn_wmma_f32_16x16x32_bf16(
                false, a, false, bm[nt], (short)0, acc[nt], false, false);
#pragma unroll
        for (int nt = 0; nt < 4; nt++) bm[nt] = wk[(nt + 4) * 32];
#pragma unroll
        for (int nt = 0; nt < 4; nt++)
            acc[4 + nt] = __builtin_amdgcn_wmma_f32_16x16x32_bf16(
                false, a, false, bm[nt], (short)0, acc[4 + nt], false, false);
    }

    // Epilogue 0: bias + ReLU + act_on(f<64, distr[0]); store bf16 -> LDS
    {
        const float d0 = distr[0], bt0 = betas[0];
#pragma unroll
        for (int nt = 0; nt < 8; nt++) {
            const int f = nt * 16 + lm;
            const float bias = b0[f];
            const bool msk = f < 64;
#pragma unroll
            for (int r = 0; r < 8; r++) {
                float h = acc[nt][r] + bias;
                h = fmaxf(h, 0.0f);
                h = act_on_f(h, msk, d0, alpha, bt0);
                sh0[(wave * 16 + r + half * 8) * 128 + f] = (__bf16)h;
            }
        }
    }

    const int rowL = wave * 16 + lm;   // A-matrix row (local) for WMMA loads from LDS

    // ================= Layer 1: 128 -> 64 =================
    v8f acc1[4];
#pragma unroll
    for (int i = 0; i < 4; i++) acc1[i] = vzero;
    {
        const __bf16* w1l = W1p + lane * 16;
#pragma unroll
        for (int ks = 0; ks < W1_KSTEPS; ks++) {
            const int kb = ks * 32;
            v16bf a;
            const __bf16* ap = &sh0[rowL * 128 + kb + kA0];
#pragma unroll
            for (int i = 0; i < 8; i++) a[i]     = ap[i];
#pragma unroll
            for (int i = 0; i < 8; i++) a[8 + i] = ap[16 + i];
            const v16bf* wk = (const v16bf*)(w1l + (size_t)ks * (W1_NT * 512));
            v16bf bm[4];
#pragma unroll
            for (int nt = 0; nt < 4; nt++) bm[nt] = wk[nt * 32];
#pragma unroll
            for (int nt = 0; nt < 4; nt++)
                acc1[nt] = __builtin_amdgcn_wmma_f32_16x16x32_bf16(
                    false, a, false, bm[nt], (short)0, acc1[nt], false, false);
        }
    }
    // Epilogue 1: +b1, act_on(d1) -> BN(g2..) -> ReLU -> act_on(d2) -> act_on(d3)
    {
        const float d1 = distr[1], bt1 = betas[1];
        const float d2 = distr[2], bt2 = betas[2];
        const float d3 = distr[3], bt3 = betas[3];
#pragma unroll
        for (int nt = 0; nt < 4; nt++) {
            const int f = nt * 16 + lm;
            const float bias = b1[f];
            const bool msk = f < 32;
            const float sc = g2[f] * rsqrtf(v2[f] + BN_EPS);
            const float mu = m2[f], bb = be2[f];
#pragma unroll
            for (int r = 0; r < 8; r++) {
                float h = acc1[nt][r] + bias;
                h = act_on_f(h, msk, d1, alpha, bt1);
                h = (h - mu) * sc + bb;
                h = fmaxf(h, 0.0f);
                h = act_on_f(h, msk, d2, alpha, bt2);
                h = act_on_f(h, msk, d3, alpha, bt3);
                sh1[(wave * 16 + r + half * 8) * 64 + f] = (__bf16)h;
            }
        }
    }

    // ================= Layer 2: 64 -> 32 =================
    v8f acc2[2];
#pragma unroll
    for (int i = 0; i < 2; i++) acc2[i] = vzero;
    {
        const __bf16* w2l = W2p + lane * 16;
#pragma unroll
        for (int ks = 0; ks < W2_KSTEPS; ks++) {
            const int kb = ks * 32;
            v16bf a;
            const __bf16* ap = &sh1[rowL * 64 + kb + kA0];
#pragma unroll
            for (int i = 0; i < 8; i++) a[i]     = ap[i];
#pragma unroll
            for (int i = 0; i < 8; i++) a[8 + i] = ap[16 + i];
            const v16bf* wk = (const v16bf*)(w2l + (size_t)ks * (W2_NT * 512));
            v16bf bm[2];
#pragma unroll
            for (int nt = 0; nt < 2; nt++) bm[nt] = wk[nt * 32];
#pragma unroll
            for (int nt = 0; nt < 2; nt++)
                acc2[nt] = __builtin_amdgcn_wmma_f32_16x16x32_bf16(
                    false, a, false, bm[nt], (short)0, acc2[nt], false, false);
        }
    }
    // Epilogue 2: +b2, act_on(d4) -> BN(g3..) -> ReLU -> act_on(d5) -> act_on(d6)
    {
        const float d4 = distr[4], bt4 = betas[4];
        const float d5 = distr[5], bt5 = betas[5];
        const float d6 = distr[6], bt6 = betas[6];
#pragma unroll
        for (int nt = 0; nt < 2; nt++) {
            const int f = nt * 16 + lm;
            const float bias = b2[f];
            const bool msk = f < 16;
            const float sc = g3[f] * rsqrtf(v3[f] + BN_EPS);
            const float mu = m3[f], bb = be3[f];
#pragma unroll
            for (int r = 0; r < 8; r++) {
                float h = acc2[nt][r] + bias;
                h = act_on_f(h, msk, d4, alpha, bt4);
                h = (h - mu) * sc + bb;
                h = fmaxf(h, 0.0f);
                h = act_on_f(h, msk, d5, alpha, bt5);
                h = act_on_f(h, msk, d6, alpha, bt6);
                sh2[(wave * 16 + r + half * 8) * 32 + f] = (__bf16)h;
            }
        }
    }

    // ================= Layer 3: 32 -> 16 (single WMMA) =================
    v8f acc3 = vzero;
    {
        v16bf a;
        const __bf16* ap = &sh2[rowL * 32 + kA0];
#pragma unroll
        for (int i = 0; i < 8; i++) a[i]     = ap[i];
#pragma unroll
        for (int i = 0; i < 8; i++) a[8 + i] = ap[16 + i];
        v16bf bm = *(const v16bf*)(W3p + lane * 16);
        acc3 = __builtin_amdgcn_wmma_f32_16x16x32_bf16(
            false, a, false, bm, (short)0, acc3, false, false);
    }

    // Epilogue 3 + Layer 4 (16 -> 1) + sigmoid, via wave32 shuffle reduction
    {
        const float d7 = distr[7], bt7 = betas[7];
        const float d8 = distr[8], bt8 = betas[8];
        const float d9 = distr[9], bt9 = betas[9];
        const int f = lm;                       // C layout: col = lane&15
        const float bias = b3[f];
        const bool msk = f < 8;
        const float sc = g4[f] * rsqrtf(v4[f] + BN_EPS);
        const float mu = m4[f], bb = be4[f];
        const float w4 = W4[f];
        float p[8];
#pragma unroll
        for (int r = 0; r < 8; r++) {
            float h = acc3[r] + bias;
            h = act_on_f(h, msk, d7, alpha, bt7);
            h = (h - mu) * sc + bb;
            h = fmaxf(h, 0.0f);
            h = act_on_f(h, msk, d8, alpha, bt8);
            h = act_on_f(h, msk, d9, alpha, bt9);
            p[r] = h * w4;
        }
        // Reduce across the 16 lanes of each half (xor masks 1,2,4,8 stay in-half)
#pragma unroll
        for (int off = 1; off < 16; off <<= 1) {
#pragma unroll
            for (int r = 0; r < 8; r++) p[r] += __shfl_xor(p[r], off, 32);
        }
        if (lm == 0) {
            const float b4v = b4[0];
#pragma unroll
            for (int r = 0; r < 8; r++) {
                float s = p[r] + b4v;
                out[rowBase + r + half * 8] = 1.0f / (1.0f + __expf(-s));
            }
        }
    }
}

extern "C" void kernel_launch(void* const* d_in, const int* in_sizes, int n_in,
                              void* d_out, int out_size, void* d_ws, size_t ws_size,
                              hipStream_t stream)
{
    (void)in_sizes; (void)n_in; (void)out_size; (void)ws_size;
    const float* x   = (const float*)d_in[0];
    const float* W0  = (const float*)d_in[1];
    const float* b0  = (const float*)d_in[2];
    const float* W1  = (const float*)d_in[3];
    const float* b1  = (const float*)d_in[4];
    const float* W2  = (const float*)d_in[5];
    const float* b2  = (const float*)d_in[6];
    const float* W3  = (const float*)d_in[7];
    const float* b3  = (const float*)d_in[8];
    const float* W4  = (const float*)d_in[9];
    const float* b4  = (const float*)d_in[10];
    const float* g2  = (const float*)d_in[11];
    const float* be2 = (const float*)d_in[12];
    const float* m2  = (const float*)d_in[13];
    const float* v2  = (const float*)d_in[14];
    const float* g3  = (const float*)d_in[15];
    const float* be3 = (const float*)d_in[16];
    const float* m3  = (const float*)d_in[17];
    const float* v3  = (const float*)d_in[18];
    const float* g4  = (const float*)d_in[19];
    const float* be4 = (const float*)d_in[20];
    const float* m4  = (const float*)d_in[21];
    const float* v4  = (const float*)d_in[22];
    // d_in[23..26] = mask128/masks64/masks32/masks16 — recomputed in-kernel (arange(d) < d/2)
    const float* distr = (const float*)d_in[27];
    const float* betas = (const float*)d_in[28];
    const float* alpha = (const float*)d_in[29];

    // Workspace layout (bf16 packed B-matrices), ~636 KB total
    __bf16* W0p = (__bf16*)d_ws;
    __bf16* W1p = W0p + W0P_ELEMS;
    __bf16* W2p = W1p + W1P_ELEMS;
    __bf16* W3p = W2p + W2P_ELEMS;

    pack_b_kernel<<<W0P_ELEMS / 256, 256, 0, stream>>>(W0, W0p, DIN, W0_NT, W0_KSTEPS);
    pack_b_kernel<<<W1P_ELEMS / 256, 256, 0, stream>>>(W1, W1p, 128, W1_NT, W1_KSTEPS);
    pack_b_kernel<<<W2P_ELEMS / 256, 256, 0, stream>>>(W2, W2p, 64,  W2_NT, W2_KSTEPS);
    pack_b_kernel<<<W3P_ELEMS / 256, 256, 0, stream>>>(W3, W3p, 32,  W3_NT, W3_KSTEPS);

    dim3 grid(BROWS / 128), block(256);
    fused_mlp<<<grid, block, 0, stream>>>(
        x, W0p, b0, W1p, b1, W2p, b2, W3p, b3, W4, b4,
        g2, be2, m2, v2, g3, be3, m3, v3, g4, be4, m4, v4,
        distr, betas, alpha, (float*)d_out);
}
